// MultiHeadAttention_1675037245638
// MI455X (gfx1250) — matrix operand order repaired and checked by reading, not compiled, not run on hardware
//
#include <hip/hip_runtime.h>

// MI455X / gfx1250 multi-head attention forward (out + attn), wave32 WMMA f16.
// 16x64 output tiling per wave: one A-tile load feeds 4 v_wmma into 4 accumulators.

typedef __attribute__((ext_vector_type(16))) _Float16 v16h;
typedef __attribute__((ext_vector_type(8)))  float    v8f;

#define HEADS 8
#define HDIM  64
#define CDIM  512
#define NTOK  1024
#define BATCH 4

// ---- WMMA tile loaders (16x16x32 f16, wave32 layouts per CDNA5 ISA 7.12.2) ----

// A: 16(M) x 32(K), source row-major f16, contiguous along K.
// lane L: m = L%16, half h = L/16; VGPR v: K = (v/4)*16 + h*8 + (v%4)*2 (+1)
static __device__ __forceinline__ v16h load_a_f16(const _Float16* __restrict__ p0,
                                                  int lda, int lane) {
  const int m = lane & 15, h = lane >> 4;
  const _Float16* p = p0 + (size_t)m * lda + h * 8;
  v16h a;
#pragma unroll
  for (int v = 0; v < 8; ++v) {
    const int k0 = (v >> 2) * 16 + (v & 3) * 2;
    a[2 * v]     = p[k0];
    a[2 * v + 1] = p[k0 + 1];
  }
  return a;
}

// Same A layout but converting from an f32 source on the fly.
static __device__ __forceinline__ v16h load_a_f32(const float* __restrict__ p0,
                                                  int lda, int lane) {
  const int m = lane & 15, h = lane >> 4;
  const float* p = p0 + (size_t)m * lda + h * 8;
  v16h a;
#pragma unroll
  for (int v = 0; v < 8; ++v) {
    const int k0 = (v >> 2) * 16 + (v & 3) * 2;
    a[2 * v]     = (_Float16)p[k0];
    a[2 * v + 1] = (_Float16)p[k0 + 1];
  }
  return a;
}

// B: 32(K) x 16(N), sourced from row-major B^T (bt[n][k]), contiguous along K.
// lane L: n = L%16, half kh = L/16; element v: K = (v/8)*16 + kh*8 + v%8
static __device__ __forceinline__ v16h load_bt_f16(const _Float16* __restrict__ bt,
                                                   int ldb, int lane) {
  const int n = lane & 15, kh = lane >> 4;
  const _Float16* p = bt + (size_t)n * ldb + kh * 8;
  v16h b;
#pragma unroll
  for (int v = 0; v < 16; ++v) b[v] = p[(v >> 3) * 16 + (v & 7)];
  return b;
}

static __device__ __forceinline__ v8f wmma16(v16h a, v16h b, v8f c) {
  return __builtin_amdgcn_wmma_f32_16x16x32_f16(false, a, false, b, (short)0, c,
                                                false, false);
}

// ---- Kernel 0: convert x to f16; transpose + convert weights to f16 ----
// idx < 2M: xh.  next 768K: qkvT (1536x512).  last 256K: projT (512x512).
__global__ __launch_bounds__(256) void prep_inputs(const float* __restrict__ x,
                                                   const float* __restrict__ qkv_w,
                                                   const float* __restrict__ proj_w,
                                                   _Float16* __restrict__ xh,
                                                   _Float16* __restrict__ qkvT,
                                                   _Float16* __restrict__ projT) {
  const int idx = blockIdx.x * 256 + threadIdx.x;
  const int NX = BATCH * NTOK * CDIM;                // 2,097,152
  if (idx < NX) {
    xh[idx] = (_Float16)x[idx];
  } else if (idx < NX + 3 * CDIM * CDIM) {
    const int j = idx - NX;                          // 1536 x 512 transpose
    const int o = j >> 9, c = j & 511;
    qkvT[j] = (_Float16)qkv_w[(size_t)c * (3 * CDIM) + o];
  } else {
    const int j = idx - NX - 3 * CDIM * CDIM;        // 512 x 512 transpose
    const int o = j >> 9, c = j & 511;
    projT[j] = (_Float16)proj_w[(size_t)c * CDIM + o];
  }
}

// ---- Kernel 1: qkv = x @ Wqkv + b; 16x64 tile/wave; scatter q(*0.125), k, v^T ----
__global__ __launch_bounds__(128) void qkv_gemm(const _Float16* __restrict__ xh,
                                                const _Float16* __restrict__ wT,
                                                const float* __restrict__ bias,
                                                _Float16* __restrict__ qws,
                                                _Float16* __restrict__ kws,
                                                _Float16* __restrict__ vTws) {
  const int lane = threadIdx.x & 31;
  const int task = blockIdx.x * 4 + (threadIdx.x >> 5);   // 256 x 24 tasks
  const int rt = task / 24, cg = task - rt * 24;
  const int rowbase = rt * 16, colbase = cg * 64;

  const _Float16* A  = xh + (size_t)rowbase * CDIM;
  const _Float16* Bt = wT + (size_t)colbase * CDIM;

  v8f acc[4] = {};
#pragma unroll 2
  for (int kk = 0; kk < CDIM; kk += 32) {
    __builtin_prefetch(A + kk + 64, 0, 1);
    const v16h a = load_a_f16(A + kk, CDIM, lane);
#pragma unroll
    for (int c = 0; c < 4; ++c)
      acc[c] = wmma16(a, load_bt_f16(Bt + (size_t)c * 16 * CDIM + kk, CDIM, lane),
                      acc[c]);
  }

  const int n = lane & 15, mh = lane >> 4;
  const int t  = colbase >> 9;              // 0:q 1:k 2:v (uniform: 64 | 512)
  const int hh = (colbase & 511) >> 6;      // head (uniform: colbase mult of 64)
#pragma unroll
  for (int c = 0; c < 4; ++c) {
    const int d = c * 16 + n;               // head-dim column
    const float bv = bias[colbase + c * 16 + n];
#pragma unroll
    for (int r = 0; r < 8; ++r) {
      const int token = rowbase + r + 8 * mh;
      const int bidx = token >> 10, nn = token & 1023;
      const float val = acc[c][r] + bv;
      const size_t qk_i = (((size_t)bidx * HEADS + hh) * NTOK + nn) * HDIM + d;
      if (t == 0)      qws[qk_i] = (_Float16)(val * 0.125f);  // fold 1/sqrt(64)
      else if (t == 1) kws[qk_i] = (_Float16)val;
      else vTws[(((size_t)bidx * HEADS + hh) * HDIM + d) * NTOK + nn] =
               (_Float16)val;
    }
  }
}

// ---- Kernel 2: scores = Q K^T (WMMA) -> LDS -> softmax -> f32 attn ----
__global__ __launch_bounds__(256) void attn_scores_softmax(
    const _Float16* __restrict__ qws, const _Float16* __restrict__ kws,
    float* __restrict__ attn) {
  __shared__ float s[16 * NTOK];            // 64 KB of the 320 KB WGP LDS
  const int lane = threadIdx.x & 31;
  const int wave = threadIdx.x >> 5;        // 8 waves
  const int qt = blockIdx.x & 63;           // query tile
  const int bh = blockIdx.x >> 6;           // b*8+h

  const _Float16* Q = qws + ((size_t)bh * NTOK + qt * 16) * HDIM;
  const int n = lane & 15, mh = lane >> 4;

  const v16h a0 = load_a_f16(Q, HDIM, lane);        // K = 0..31 of D
  const v16h a1 = load_a_f16(Q + 32, HDIM, lane);   // K = 32..63

#pragma unroll 1
  for (int j = 0; j < 8; ++j) {                      // 8 key tiles per wave
    const int ctile = wave + 8 * j;
    const _Float16* Kt = kws + ((size_t)bh * NTOK + ctile * 16) * HDIM;
    v8f acc = {};
    acc = wmma16(a0, load_bt_f16(Kt, HDIM, lane), acc);
    acc = wmma16(a1, load_bt_f16(Kt + 32, HDIM, lane), acc);
    const int col = ctile * 16 + n;
#pragma unroll
    for (int r = 0; r < 8; ++r) s[(r + 8 * mh) * NTOK + col] = acc[r];
  }
  __syncthreads();

  // Softmax: each wave owns 2 rows; 16 lanes per row, 64 elements per lane.
  const int row = 2 * wave + mh;
  float* srow = s + row * NTOK;
  float mx = -3.4e38f;
#pragma unroll 4
  for (int j = 0; j < 64; ++j) mx = fmaxf(mx, srow[n + 16 * j]);
#pragma unroll
  for (int m = 1; m < 16; m <<= 1) mx = fmaxf(mx, __shfl_xor(mx, m, 32));
  float sum = 0.f;
#pragma unroll 4
  for (int j = 0; j < 64; ++j) {
    const float e = __expf(srow[n + 16 * j] - mx);
    srow[n + 16 * j] = e;
    sum += e;
  }
#pragma unroll
  for (int m = 1; m < 16; m <<= 1) sum += __shfl_xor(sum, m, 32);
  const float inv = 1.0f / sum;
  float* outp = attn + ((size_t)bh * NTOK + qt * 16 + row) * NTOK;
#pragma unroll 4
  for (int j = 0; j < 64; ++j) outp[n + 16 * j] = srow[n + 16 * j] * inv;
}

// ---- Kernel 3: ctx = attn @ V; one wave does full 16x64 tile (1 cvt pass) ----
__global__ __launch_bounds__(128) void attn_v(const float* __restrict__ attn,
                                              const _Float16* __restrict__ vT,
                                              _Float16* __restrict__ ctx) {
  const int lane = threadIdx.x & 31;
  const int task = blockIdx.x * 4 + (threadIdx.x >> 5);   // 32 x 64 tasks
  const int qt = task & 63;
  const int bh = task >> 6;

  const float*    A  = attn + ((size_t)bh * NTOK + qt * 16) * NTOK;
  const _Float16* Bt = vT + (size_t)bh * HDIM * NTOK;

  v8f acc[4] = {};
#pragma unroll 2
  for (int kk = 0; kk < NTOK; kk += 32) {
    __builtin_prefetch(A + kk + 64, 0, 1);
    const v16h a = load_a_f32(A + kk, NTOK, lane);    // attn in [0,1] -> f16 safe
#pragma unroll
    for (int c = 0; c < 4; ++c)
      acc[c] = wmma16(a, load_bt_f16(Bt + (size_t)c * 16 * NTOK + kk, NTOK, lane),
                      acc[c]);
  }

  const int n = lane & 15, mh = lane >> 4;
  const int b_ = bh >> 3, h_ = bh & 7;
#pragma unroll
  for (int c = 0; c < 4; ++c)
#pragma unroll
    for (int r = 0; r < 8; ++r) {
      const int tok = qt * 16 + r + 8 * mh;
      ctx[((size_t)b_ * NTOK + tok) * CDIM + h_ * HDIM + c * 16 + n] =
          (_Float16)acc[c][r];
    }
}

// ---- Kernel 4: out = ctx @ Wproj + b  (f32 output); 16x64 tile/wave ----
__global__ __launch_bounds__(128) void proj_gemm(const _Float16* __restrict__ ctx,
                                                 const _Float16* __restrict__ wT,
                                                 const float* __restrict__ bias,
                                                 float* __restrict__ out) {
  const int lane = threadIdx.x & 31;
  const int task = blockIdx.x * 4 + (threadIdx.x >> 5);   // 256 x 8 tasks
  const int rt = task >> 3, cg = task & 7;
  const int rowbase = rt * 16, colbase = cg * 64;
  const _Float16* A  = ctx + (size_t)rowbase * CDIM;
  const _Float16* Bt = wT  + (size_t)colbase * CDIM;

  v8f acc[4] = {};
#pragma unroll 2
  for (int kk = 0; kk < CDIM; kk += 32) {
    __builtin_prefetch(A + kk + 64, 0, 1);
    const v16h a = load_a_f16(A + kk, CDIM, lane);
#pragma unroll
    for (int c = 0; c < 4; ++c)
      acc[c] = wmma16(a, load_bt_f16(Bt + (size_t)c * 16 * CDIM + kk, CDIM, lane),
                      acc[c]);
  }

  const int n = lane & 15, mh = lane >> 4;
#pragma unroll
  for (int c = 0; c < 4; ++c) {
    const float bv = bias[colbase + c * 16 + n];
#pragma unroll
    for (int r = 0; r < 8; ++r) {
      const int tok = rowbase + r + 8 * mh;
      out[(size_t)tok * CDIM + colbase + c * 16 + n] = acc[c][r] + bv;
    }
  }
}

extern "C" void kernel_launch(void* const* d_in, const int* in_sizes, int n_in,
                              void* d_out, int out_size, void* d_ws, size_t ws_size,
                              hipStream_t stream) {
  const float* x      = (const float*)d_in[0];
  const float* qkv_w  = (const float*)d_in[1];
  const float* qkv_b  = (const float*)d_in[2];
  const float* proj_w = (const float*)d_in[3];
  const float* proj_b = (const float*)d_in[4];

  float* out  = (float*)d_out;                         // [4,1024,512]
  float* attn = out + (size_t)BATCH * NTOK * CDIM;     // [4,8,1024,1024]

  _Float16* ws    = (_Float16*)d_ws;                   // ~22 MB total
  _Float16* xh    = ws; ws += (size_t)BATCH * NTOK * CDIM; // x as f16
  _Float16* qkvT  = ws; ws += (size_t)3 * CDIM * CDIM;     // 1536x512
  _Float16* projT = ws; ws += (size_t)CDIM * CDIM;         // 512x512
  _Float16* qws   = ws; ws += (size_t)BATCH * NTOK * CDIM; // [b,h,n,d]
  _Float16* kws   = ws; ws += (size_t)BATCH * NTOK * CDIM; // [b,h,n,d]
  _Float16* vT    = ws; ws += (size_t)BATCH * NTOK * CDIM; // [b,h,d,n]
  _Float16* ctx   = ws;                                    // [b,n,c]

  hipLaunchKernelGGL(prep_inputs, dim3(12288), dim3(256), 0, stream,
                     x, qkv_w, proj_w, xh, qkvT, projT);
  hipLaunchKernelGGL(qkv_gemm, dim3(1536), dim3(128), 0, stream,
                     xh, qkvT, qkv_b, qws, kws, vT);
  hipLaunchKernelGGL(attn_scores_softmax, dim3(2048), dim3(256), 0, stream,
                     qws, kws, attn);
  hipLaunchKernelGGL(attn_v, dim3(512), dim3(128), 0, stream, attn, vT, ctx);
  hipLaunchKernelGGL(proj_gemm, dim3(512), dim3(128), 0, stream,
                     ctx, projT, proj_b, out);
}
